// Block_24283745092155
// MI455X (gfx1250) — matrix operand order repaired and checked by reading, not compiled, not run on hardware
//
#include <hip/hip_runtime.h>
#include <hip/hip_bf16.h>

// ---------------------------------------------------------------------------
// Mamba block for MI455X (gfx1250): bf16 WMMA GEMMs + scalarized selective scan
// ---------------------------------------------------------------------------

typedef __attribute__((ext_vector_type(16))) __bf16 v16bf;
typedef __attribute__((ext_vector_type(8)))  __bf16 v8bf;
typedef __attribute__((ext_vector_type(8)))  float  v8f;

#define D_MODEL 1024
#define D_STATE 16
#define D_CONV  4
#define D_INNER 2048
#define DT_RANK 64
#define BATCH   2
#define SEQL    2048
#define MROWS   (BATCH * SEQL)   // 4096

// ---------------------------------------------------------------------------
// f32 -> bf16 cast (grid-stride)
// ---------------------------------------------------------------------------
__global__ void cast_f32_bf16_kernel(const float* __restrict__ in,
                                     __bf16* __restrict__ out, int n) {
  for (int i = blockIdx.x * blockDim.x + threadIdx.x; i < n;
       i += gridDim.x * blockDim.x)
    out[i] = (__bf16)in[i];
}

// Extract dt_low (first DT_RANK cols of x_dbl [M,96]) as bf16 [M,64]
__global__ void cast_dtlow_kernel(const float* __restrict__ xdbl,
                                  __bf16* __restrict__ dtl) {
  int g = blockIdx.x * blockDim.x + threadIdx.x;
  if (g >= MROWS * DT_RANK) return;
  int m = g >> 6, c = g & 63;
  dtl[g] = (__bf16)xdbl[(size_t)m * 96 + c];
}

// ---------------------------------------------------------------------------
// LayerNorm: one block per row of 1024, output bf16
// ---------------------------------------------------------------------------
__global__ __launch_bounds__(256)
void layernorm_kernel(const float* __restrict__ x, const float* __restrict__ w,
                      const float* __restrict__ b, __bf16* __restrict__ y) {
  const int row = blockIdx.x;
  const float* xr = x + (size_t)row * D_MODEL;
  float s = 0.f, s2 = 0.f;
  for (int i = threadIdx.x; i < D_MODEL; i += 256) {
    float v = xr[i];
    s += v; s2 += v * v;
  }
  #pragma unroll
  for (int off = 16; off > 0; off >>= 1) {
    s  += __shfl_xor(s,  off, 32);
    s2 += __shfl_xor(s2, off, 32);
  }
  __shared__ float ps[8], ps2[8];
  int wv = threadIdx.x >> 5, ln = threadIdx.x & 31;
  if (ln == 0) { ps[wv] = s; ps2[wv] = s2; }
  __syncthreads();
  if (threadIdx.x == 0) {
    float a = 0.f, a2 = 0.f;
    #pragma unroll
    for (int i = 0; i < 8; i++) { a += ps[i]; a2 += ps2[i]; }
    ps[0] = a; ps2[0] = a2;
  }
  __syncthreads();
  const float mean = ps[0] * (1.f / D_MODEL);
  const float var  = ps2[0] * (1.f / D_MODEL) - mean * mean;
  const float rs   = rsqrtf(var + 1e-5f);
  __bf16* yr = y + (size_t)row * D_MODEL;
  for (int i = threadIdx.x; i < D_MODEL; i += 256)
    yr[i] = (__bf16)((xr[i] - mean) * rs * w[i] + b[i]);
}

// ---------------------------------------------------------------------------
// Generic C = A * W^T GEMM on the bf16 WMMA pipe.
//   A: [M,K] bf16 row-major (activations)
//   W: [N,K] bf16 row-major (torch-style weight)
// Block: 256 threads = 8 waves stacked along M (128 rows); each wave computes
// a 16(M) x 16*NT(N) strip = NT accumulators, reusing its A fragment NT x.
// NT is a compile-time constant so every WMMA in the K-loop is UNCONDITIONAL
// (EXEC all-1s, no accumulator phi copies).  All fragment loads for a K-step
// are issued BEFORE any WMMA consumes them, so they land in distinct VGPRs
// and the hardware can overlap the load clause with the WMMA sequence.
// Launch grids are exact: grid = (M/128, N/(16*NT)).
// A/B k order: contiguous, lane half h holds K = 16h+0..15 (same in A and B); f32 C 16x16 per ISA 7.12.2.
// EPI: 0 = store bf16 | 1 = store f32 | 2 = softplus(acc+bias) f32
//      3 = acc + residual, f32
// ---------------------------------------------------------------------------
template <int EPI, int NT>
__global__ __launch_bounds__(256)
void gemm_wmma_kernel(const __bf16* __restrict__ A, const __bf16* __restrict__ W,
                      int K,
                      float* __restrict__ outF, __bf16* __restrict__ outB,
                      const float* __restrict__ bias,
                      const float* __restrict__ resid, int ldo) {
  const int lane = threadIdx.x & 31;
  const int wave = threadIdx.x >> 5;
  const int r    = lane & 15;   // N / M-row selector within half-wave
  const int hi   = lane >> 4;   // half-wave selector

  const int m0 = blockIdx.x * 128 + wave * 16;
  const int n0 = blockIdx.y * (16 * NT);

  v8f acc[NT] = {};

  const __bf16* aRow = A + (size_t)(m0 + r) * K + hi * 16;
  const __bf16* wRow[NT];
  #pragma unroll
  for (int i = 0; i < NT; i++)
    wRow[i] = W + (size_t)(n0 + i * 16 + r) * K + hi * 16;

  #pragma unroll 2
  for (int k0 = 0; k0 < K; k0 += 32) {
    // ---- issue all loads for this K-step first (distinct registers) ----
    // A fragment (16x32): lanes 0-15 hold K = k0+{0..15},
    //                     lanes 16-31 hold K = k0+{16..31}
    const __bf16* ap = aRow + k0;
    __builtin_prefetch(ap + 64, 0, 3);  // next K-steps -> global_prefetch_b8
    v8bf a_lo = *(const v8bf*)(ap);
    v8bf a_hi = *(const v8bf*)(ap + 8);
    v16bf bfrag[NT];
    #pragma unroll
    for (int i = 0; i < NT; i++)
      // B fragment (32x16): lane's column = n0+16i+r, K span = k0 + hi*16 + 0..15
      bfrag[i] = *(const v16bf*)(wRow[i] + k0);

    v16bf afrag;
    #pragma unroll
    for (int j = 0; j < 8; j++) { afrag[j] = a_lo[j]; afrag[8 + j] = a_hi[j]; }

    // ---- consume: NT back-to-back WMMAs ----
    #pragma unroll
    for (int i = 0; i < NT; i++)
      acc[i] = __builtin_amdgcn_wmma_f32_16x16x32_bf16(
          false, afrag, false, bfrag[i], (short)0, acc[i], false, false);
  }

  // C layout: VGPR j -> row m0 + hi*8 + j, col n0 + 16i + r
  #pragma unroll
  for (int i = 0; i < NT; i++) {
    const int col = n0 + i * 16 + r;
    #pragma unroll
    for (int j = 0; j < 8; j++) {
      const int row = m0 + hi * 8 + j;
      float v = acc[i][j];
      const size_t o = (size_t)row * ldo + col;
      if (EPI == 0) {
        outB[o] = (__bf16)v;
      } else if (EPI == 1) {
        outF[o] = v;
      } else if (EPI == 2) {
        float t = v + bias[col];
        outF[o] = (t > 20.f) ? t : log1pf(__expf(t));
      } else {
        outF[o] = v + resid[o];
      }
    }
  }
}

// ---------------------------------------------------------------------------
// Depthwise causal conv (k=4) + bias + SiLU.  u lives in xz[:, 0:2048] (bf16,
// row stride 4096).  Output bf16 u_c [M, 2048].
// ---------------------------------------------------------------------------
__global__ void conv_silu_kernel(const __bf16* __restrict__ xz,
                                 const float* __restrict__ cw,
                                 const float* __restrict__ cb,
                                 __bf16* __restrict__ ucb) {
  int g = blockIdx.x * blockDim.x + threadIdx.x;
  if (g >= MROWS * D_INNER) return;
  const int m = g >> 11;        // b*L + l
  const int d = g & 2047;
  const int l = m & (SEQL - 1);
  float acc = cb[d];
  #pragma unroll
  for (int j = 0; j < D_CONV; j++) {
    const int li = l - (D_CONV - 1) + j;
    if (li >= 0)
      acc += (float)xz[(size_t)(m - (D_CONV - 1) + j) * (2 * D_INNER) + d] *
             cw[d * D_CONV + j];
  }
  const float sg = 1.f / (1.f + __expf(-acc));
  ucb[g] = (__bf16)(acc * sg);
}

// ---------------------------------------------------------------------------
// Selective scan.  One thread per (b, d) channel: h[16] and A[16] in VGPRs,
// sequential over t.  B/C addresses are block-uniform (blockIdx + t only), so
// they lower to scalar loads.  ys written in place over dt.
// ---------------------------------------------------------------------------
__global__ __launch_bounds__(256)
void scan_kernel(const float* __restrict__ xdbl, const __bf16* __restrict__ ucb,
                 const float* __restrict__ A_log, float* __restrict__ dt) {
  const int gbase = blockIdx.x * 256;           // 2048 % 256 == 0 -> b uniform
  const int b = gbase >> 11;
  const int d = (gbase & 2047) + threadIdx.x;

  float Arow[D_STATE], h[D_STATE];
  #pragma unroll
  for (int n = 0; n < D_STATE; n++) {
    Arow[n] = -__expf(A_log[(size_t)d * D_STATE + n]);
    h[n] = 0.f;
  }
  const float* bcBase = xdbl + (size_t)b * SEQL * 96 + DT_RANK;  // uniform
  size_t idx = (size_t)b * SEQL * D_INNER + d;

  for (int t = 0; t < SEQL; t++) {
    const float* bc = bcBase + (size_t)t * 96;  // uniform -> s_load
    const float dtv = dt[idx];
    const float du  = dtv * (float)ucb[idx];
    float y = 0.f;
    #pragma unroll
    for (int n = 0; n < D_STATE; n++) {
      const float dA = __expf(dtv * Arow[n]);
      h[n] = dA * h[n] + du * bc[n];            // B
      y += h[n] * bc[D_STATE + n];              // C
    }
    dt[idx] = y;                                // ys in place
    idx += D_INNER;
  }
}

// ---------------------------------------------------------------------------
// Gate: ysb = (ys + u_c * Dskip) * silu(z), bf16 output for out_proj.
// ---------------------------------------------------------------------------
__global__ void gate_kernel(const float* __restrict__ ys,
                            const __bf16* __restrict__ ucb,
                            const float* __restrict__ Dskip,
                            const __bf16* __restrict__ xz,
                            __bf16* __restrict__ ysb) {
  int g = blockIdx.x * blockDim.x + threadIdx.x;
  if (g >= MROWS * D_INNER) return;
  const int m = g >> 11, d = g & 2047;
  float v = ys[g] + (float)ucb[g] * Dskip[d];
  const float z = (float)xz[(size_t)m * (2 * D_INNER) + D_INNER + d];
  v *= z / (1.f + __expf(-z));
  ysb[g] = (__bf16)v;
}

// ---------------------------------------------------------------------------
// Host-side launch
// ---------------------------------------------------------------------------
static inline size_t carve(size_t& off, size_t bytes) {
  size_t o = off;
  off = (off + bytes + 255) & ~(size_t)255;
  return o;
}

extern "C" void kernel_launch(void* const* d_in, const int* in_sizes, int n_in,
                              void* d_out, int out_size, void* d_ws, size_t ws_size,
                              hipStream_t stream) {
  const float* x        = (const float*)d_in[0];
  const float* ln_w     = (const float*)d_in[1];
  const float* ln_b     = (const float*)d_in[2];
  const float* in_proj  = (const float*)d_in[3];
  const float* conv_w   = (const float*)d_in[4];
  const float* conv_b   = (const float*)d_in[5];
  const float* x_proj   = (const float*)d_in[6];
  const float* dt_proj  = (const float*)d_in[7];
  const float* dt_bias  = (const float*)d_in[8];
  const float* A_log    = (const float*)d_in[9];
  const float* Dskip    = (const float*)d_in[10];
  const float* out_proj = (const float*)d_in[11];
  float* out = (float*)d_out;

  char* ws = (char*)d_ws;
  size_t off = 0;
  __bf16* ybf   = (__bf16*)(ws + carve(off, (size_t)MROWS * D_MODEL * 2));        // 8 MB
  __bf16* xz    = (__bf16*)(ws + carve(off, (size_t)MROWS * 2 * D_INNER * 2));    // 32 MB
  __bf16* ucb   = (__bf16*)(ws + carve(off, (size_t)MROWS * D_INNER * 2));        // 16 MB
  float*  xdbl  = (float*) (ws + carve(off, (size_t)MROWS * 96 * 4));             // 1.5 MB
  __bf16* dtl   = (__bf16*)(ws + carve(off, (size_t)MROWS * DT_RANK * 2));        // 0.5 MB
  float*  dtf   = (float*) (ws + carve(off, (size_t)MROWS * D_INNER * 4));        // 32 MB (dt, then ys)
  __bf16* ysb   = (__bf16*)(ws + carve(off, (size_t)MROWS * D_INNER * 2));        // 16 MB
  __bf16* w_in  = (__bf16*)(ws + carve(off, (size_t)2 * D_INNER * D_MODEL * 2));  // 8 MB
  __bf16* w_xp  = (__bf16*)(ws + carve(off, (size_t)96 * D_INNER * 2));
  __bf16* w_dt  = (__bf16*)(ws + carve(off, (size_t)D_INNER * DT_RANK * 2));
  __bf16* w_out = (__bf16*)(ws + carve(off, (size_t)D_MODEL * D_INNER * 2));      // 4 MB
  (void)ws_size;

  const dim3 blk(256);
  auto gs = [](size_t n) { return dim3((unsigned)((n + 255) / 256)); };

  // weight casts (L2-resident afterwards)
  cast_f32_bf16_kernel<<<gs((size_t)2 * D_INNER * D_MODEL), blk, 0, stream>>>(in_proj,  w_in,  2 * D_INNER * D_MODEL);
  cast_f32_bf16_kernel<<<gs((size_t)96 * D_INNER),          blk, 0, stream>>>(x_proj,   w_xp,  96 * D_INNER);
  cast_f32_bf16_kernel<<<gs((size_t)D_INNER * DT_RANK),     blk, 0, stream>>>(dt_proj,  w_dt,  D_INNER * DT_RANK);
  cast_f32_bf16_kernel<<<gs((size_t)D_MODEL * D_INNER),     blk, 0, stream>>>(out_proj, w_out, D_MODEL * D_INNER);

  // 1) pre-norm
  layernorm_kernel<<<dim3(MROWS), blk, 0, stream>>>(x, ln_w, ln_b, ybf);

  // 2) in_proj: [4096,1024] x [4096,1024]^T -> xz bf16 [4096,4096]
  gemm_wmma_kernel<0, 4><<<dim3(MROWS / 128, (2 * D_INNER) / 64), blk, 0, stream>>>(
      ybf, w_in, D_MODEL, nullptr, xz, nullptr, nullptr, 2 * D_INNER);

  // 3) depthwise causal conv + SiLU -> u_c bf16
  conv_silu_kernel<<<gs((size_t)MROWS * D_INNER), blk, 0, stream>>>(xz, conv_w, conv_b, ucb);

  // 4) x_proj: [4096,2048] x [96,2048]^T -> x_dbl f32 [4096,96]  (96 = 3 x 32)
  gemm_wmma_kernel<1, 2><<<dim3(MROWS / 128, 3), blk, 0, stream>>>(
      ucb, w_xp, D_INNER, xdbl, nullptr, nullptr, nullptr, 96);

  // 5) dt_proj on low-rank dt: softplus(dtl @ W^T + b) -> dt f32 [4096,2048]
  cast_dtlow_kernel<<<gs((size_t)MROWS * DT_RANK), blk, 0, stream>>>(xdbl, dtl);
  gemm_wmma_kernel<2, 4><<<dim3(MROWS / 128, D_INNER / 64), blk, 0, stream>>>(
      dtl, w_dt, DT_RANK, dtf, nullptr, dt_bias, nullptr, D_INNER);

  // 6) selective scan (ys overwrites dt in place)
  scan_kernel<<<dim3((BATCH * D_INNER) / 256), blk, 0, stream>>>(xdbl, ucb, A_log, dtf);

  // 7) skip + gate -> bf16
  gate_kernel<<<gs((size_t)MROWS * D_INNER), blk, 0, stream>>>(dtf, ucb, Dskip, xz, ysb);

  // 8) out_proj + residual -> d_out f32 [4096,1024]
  gemm_wmma_kernel<3, 4><<<dim3(MROWS / 128, D_MODEL / 64), blk, 0, stream>>>(
      ysb, w_out, D_INNER, out, nullptr, nullptr, x, D_MODEL);
  (void)in_sizes; (void)n_in; (void)out_size;
}
